// TransformerModel_91096256348660
// MI455X (gfx1250) — matrix operand order repaired and checked by reading, not compiled, not run on hardware
//
#include <hip/hip_runtime.h>
#include <hip/hip_fp16.h>
#include <math.h>

// ---------------------------------------------------------------------------
// MI455X (gfx1250) transformer forward.
// Compute-bound (~473 GFLOP vs ~15us memory floor at 23.3 TB/s): all GEMMs and
// attention run on v_wmma_f32_16x16x32_f16 (f16 in, f32 accumulate).
// Pure-copy LDS stagings use GLOBAL_LOAD_ASYNC_TO_LDS_B128 (ASYNCcnt).
// Flash-style attention avoids materializing the 268MB score tensor.
// ---------------------------------------------------------------------------

typedef __attribute__((ext_vector_type(16))) _Float16 v16h;
typedef __attribute__((ext_vector_type(8)))  float    v8f;

union FragH { v16h v; uint4 q[2]; };
union PackH { _Float16 h[2]; unsigned u; };

#define SEQ   2048
#define DM    1024
#define NHEAD 16
#define DHEAD 64
#define DHID  4096
#define NLAY  6
#define NTOK  2049

// ---- CDNA5 async copy: global -> LDS, 16B per lane, tracked by ASYNCcnt ----
static __device__ __forceinline__ void async_g2l_b128(void* ldsDst, const void* gSrc) {
    asm volatile("global_load_async_to_lds_b128 %0, %1, off"
                 :: "v"((unsigned)(size_t)ldsDst),
                    "v"((unsigned long long)(size_t)gSrc)
                 : "memory");
}
static __device__ __forceinline__ void wait_async0() {
    asm volatile("s_wait_asynccnt 0x0" ::: "memory");
}

static __device__ __forceinline__ float red16_max(float v) {
    v = fmaxf(v, __shfl_xor(v, 1, 32));
    v = fmaxf(v, __shfl_xor(v, 2, 32));
    v = fmaxf(v, __shfl_xor(v, 4, 32));
    v = fmaxf(v, __shfl_xor(v, 8, 32));
    return v;
}
static __device__ __forceinline__ float red16_sum(float v) {
    v += __shfl_xor(v, 1, 32);
    v += __shfl_xor(v, 2, 32);
    v += __shfl_xor(v, 4, 32);
    v += __shfl_xor(v, 8, 32);
    return v;
}

// ---------------------------------------------------------------------------
// WMMA GEMM: C[M,N] = act( A_h[M,Kloop](f16) * W[N,Kw](f32)^T + bias [+res] )
// Block tile 128x128, BK=64, 256 threads = 8 waves (2 M x 4 N); wave tile
// 64x32 -> 16 WMMA per staging interval, 8 f32 accumulators.
// Requires: Kloop % 64 == 0, lda % 8 == 0, Kw even, M % 128 == 0.
// Columns in [N, nWrite) are written as zeros (for padded f16 buffers).
// ---------------------------------------------------------------------------
__global__ void __launch_bounds__(256)
gemm_wmma(const _Float16* __restrict__ A, int lda,
          const float* __restrict__ W, int Kloop, int Kw, int N,
          const float* __restrict__ bias,
          const float* __restrict__ resid,        // [M,N] ld=N, or nullptr
          float* __restrict__ outF, int ldc,      // nullptr to skip
          _Float16* __restrict__ outH, int ldh,   // nullptr to skip
          int relu, int nWrite)
{
    __shared__ _Float16 As[128][88];    // 64 used + pad, rows 176B (16B mult)
    __shared__ _Float16 Bs[64][152];    // 128 used + pad, rows 304B (16B mult)

    const int t    = threadIdx.x;
    const int lane = t & 31;
    const int wid  = t >> 5;
    const int wm   = wid >> 2;           // 0..1
    const int wn   = wid & 3;            // 0..3
    const int mBlock = blockIdx.y * 128;
    const int nBlock = blockIdx.x * 128;

    v8f zero = {};
    v8f acc[4][2];
#pragma unroll
    for (int mi = 0; mi < 4; ++mi)
#pragma unroll
        for (int ni = 0; ni < 2; ++ni) acc[mi][ni] = zero;

    const int kSteps = Kloop / 64;
    for (int ks = 0; ks < kSteps; ++ks) {
        const int kb = ks * 64;
        __syncthreads();
        // ---- stage A tile (128x64 f16) via async global->LDS ----
        {
            const int c  = (t & 7) * 8;
            const int r0 = t >> 3;
#pragma unroll
            for (int rr = 0; rr < 4; ++rr) {
                const int r = r0 + rr * 32;
                async_g2l_b128(&As[r][c],
                               A + (size_t)(mBlock + r) * lda + kb + c);
            }
        }
        // ---- stage B tile: transpose W[N,Kw] fp32 -> Bs[k][n] f16 ----
        {
            const int p  = t >> 2;            // n pair 0..63 -> n = 2p, 2p+1
            const int k0 = (t & 3) * 16;      // k chunk of 16
            const int gn0 = nBlock + 2 * p;
            const int gn1 = gn0 + 1;
            const float* w0 = W + (size_t)gn0 * Kw + kb + k0;
            const float* w1 = W + (size_t)gn1 * Kw + kb + k0;
            const bool v0 = gn0 < N, v1 = gn1 < N;
            if (v0 && kb + k0 + 80 < Kw)
                __builtin_prefetch(w0 + 64, 0, 1);   // next K tile of this row
#pragma unroll
            for (int j = 0; j < 8; ++j) {
                const int k = kb + k0 + 2 * j;       // Kw even: pair never straddles
                const bool kv = k < Kw;
                float2 a = (v0 && kv) ? *(const float2*)(w0 + 2 * j)
                                      : make_float2(0.f, 0.f);
                float2 b = (v1 && kv) ? *(const float2*)(w1 + 2 * j)
                                      : make_float2(0.f, 0.f);
                PackH p0, p1;
                p0.h[0] = (_Float16)a.x; p0.h[1] = (_Float16)b.x;
                p1.h[0] = (_Float16)a.y; p1.h[1] = (_Float16)b.y;
                *(unsigned*)(&Bs[k0 + 2 * j][2 * p])     = p0.u;
                *(unsigned*)(&Bs[k0 + 2 * j + 1][2 * p]) = p1.u;
            }
        }
        wait_async0();
        __syncthreads();
        // ---- compute: 16 WMMA per wave per interval ----
#pragma unroll
        for (int kk = 0; kk < 64; kk += 32) {
            FragH fb[2];
            const int n0 = wn * 32;
#pragma unroll
            for (int ni = 0; ni < 2; ++ni) {
#pragma unroll
                for (int s = 0; s < 16; ++s)
                    fb[ni].v.data()[0], fb[ni].v[s] = Bs[kk + ((s < 8) ? 0 : 16) + ((lane < 16) ? 0 : 8) + (s & 7)][n0 + ni * 16 + (lane & 15)];
            }
#pragma unroll
            for (int mi = 0; mi < 4; ++mi) {
                FragH fa;
                const int m    = wm * 64 + mi * 16 + (lane & 15);
                const int kOff = kk + ((lane < 16) ? 0 : 8);
                fa.q[0] = *(const uint4*)(&As[m][kOff]);
                fa.q[1] = *(const uint4*)(&As[m][kOff + 16]);
#pragma unroll
                for (int ni = 0; ni < 2; ++ni) {
                    acc[mi][ni] = __builtin_amdgcn_wmma_f32_16x16x32_f16(
                        false, fa.v, false, fb[ni].v, (short)0, acc[mi][ni],
                        false, false);
                }
            }
        }
    }

    // ---- epilogue ----
#pragma unroll
    for (int ni = 0; ni < 2; ++ni) {
        const int col = nBlock + wn * 32 + ni * 16 + (lane & 15);
        if (col >= nWrite) continue;
        const bool valid = col < N;
        const float bval = valid ? bias[col] : 0.0f;
#pragma unroll
        for (int mi = 0; mi < 4; ++mi) {
#pragma unroll
            for (int e = 0; e < 8; ++e) {
                const int row = mBlock + wm * 64 + mi * 16 + ((lane < 16) ? e : e + 8);
                float v = 0.0f;
                if (valid) {
                    v = acc[mi][ni][e] + bval;
                    if (resid) v += resid[(size_t)row * N + col];
                    if (relu)  v = fmaxf(v, 0.0f);
                }
                if (outF) outF[(size_t)row * ldc + col] = v;
                if (outH) outH[(size_t)row * ldh + col] = (_Float16)v;
            }
        }
    }
}

// ---------------------------------------------------------------------------
// Fused flash attention. One block per (128 query rows, head).
// 8 waves; wave w owns query rows [w*16, w*16+16). Streams keys in blocks of
// 32, online softmax, QK^T and PV both on WMMA. Q/V staged with async copies.
// qkv layout: [S, 3*DM] f16 (q | k | v per row, head-contiguous 64-chunks).
// ---------------------------------------------------------------------------
__global__ void __launch_bounds__(256)
attn_wmma(const _Float16* __restrict__ qkv, _Float16* __restrict__ ctx)
{
    __shared__ _Float16 Qs[128][80];        // 128 q rows x 64 dims (+pad)
    __shared__ _Float16 Kts[64][48];        // dims x 32 keys (+pad), transposed
    __shared__ _Float16 Vs[32][80];         // 32 keys x 64 dims (+pad)
    __shared__ _Float16 Ps[8][16][48];      // per-wave P transpose scratch

    const int t    = threadIdx.x;
    const int lane = t & 31;
    const int wid  = t >> 5;
    const int h    = blockIdx.y;
    const int qBlock = blockIdx.x * 128;
    const int ldq  = 3 * DM;

    // stage Q block once (async; completion covered by first wait_async0)
    {
        const int r = t >> 1;
        const int c = (t & 1) * 32;
        const _Float16* src = qkv + (size_t)(qBlock + r) * ldq + h * DHEAD + c;
#pragma unroll
        for (int j = 0; j < 4; ++j)
            async_g2l_b128(&Qs[r][c + j * 8], src + j * 8);
    }

    v8f zero = {};
    v8f acc[4];
#pragma unroll
    for (int i = 0; i < 4; ++i) acc[i] = zero;
    float mr[8], lr[8];
#pragma unroll
    for (int e = 0; e < 8; ++e) { mr[e] = -1e30f; lr[e] = 0.0f; }

    _Float16* PsW = &Ps[wid][0][0];

    for (int kb = 0; kb < SEQ; kb += 32) {
        __syncthreads();
        // stage V tile async: Vs[key][dim]
        {
            const int kl = t >> 3;
            const int d0 = (t & 7) * 8;
            async_g2l_b128(&Vs[kl][d0],
                           qkv + (size_t)(kb + kl) * ldq + 2 * DM + h * DHEAD + d0);
        }
        // stage K^T tile (transpose, VGPR path): Kts[dim][key]
        {
            const int kl = t & 31;
            const int d0 = (t >> 5) * 8;
            const _Float16* src = qkv + (size_t)(kb + kl) * ldq + DM + h * DHEAD + d0;
            if (kb + 32 < SEQ)
                __builtin_prefetch(src + 32 * ldq, 0, 1);
            union { uint4 q; _Float16 h8[8]; } u;
            u.q = *(const uint4*)src;
#pragma unroll
            for (int j = 0; j < 8; ++j) Kts[d0 + j][kl] = u.h8[j];
        }
        wait_async0();
        __syncthreads();

        // ---- scores: 16 q rows x 32 keys, contraction over 64 dims ----
        v8f s0 = zero, s1 = zero;
#pragma unroll
        for (int j = 0; j < 2; ++j) {
            FragH fa, fb0, fb1;
            const int m    = wid * 16 + (lane & 15);
            const int kOff = j * 32 + ((lane < 16) ? 0 : 8);
            fa.q[0] = *(const uint4*)(&Qs[m][kOff]);
            fa.q[1] = *(const uint4*)(&Qs[m][kOff + 16]);
#pragma unroll
            for (int s = 0; s < 16; ++s) {
                const int dr = j * 32 + ((s < 8) ? 0 : 16) + ((lane < 16) ? 0 : 8) + (s & 7);
                fb0.v[s] = Kts[dr][lane & 15];
                fb1.v[s] = Kts[dr][16 + (lane & 15)];
            }
            s0 = __builtin_amdgcn_wmma_f32_16x16x32_f16(false, fa.v, false, fb0.v,
                                                        (short)0, s0, false, false);
            s1 = __builtin_amdgcn_wmma_f32_16x16x32_f16(false, fa.v, false, fb1.v,
                                                        (short)0, s1, false, false);
        }

        // ---- online softmax over these 32 keys ----
#pragma unroll
        for (int e = 0; e < 8; ++e) {
            const float a0 = s0[e] * 0.125f;              // 1/sqrt(64)
            const float a1 = s1[e] * 0.125f;
            const float mx   = red16_max(fmaxf(a0, a1));
            const float mnew = fmaxf(mr[e], mx);
            const float corr = __expf(mr[e] - mnew);
            const float p0 = __expf(a0 - mnew);
            const float p1 = __expf(a1 - mnew);
            const float rs = red16_sum(p0 + p1);
            lr[e] = lr[e] * corr + rs;
            mr[e] = mnew;
            acc[0][e] *= corr; acc[1][e] *= corr;
            acc[2][e] *= corr; acc[3][e] *= corr;
            const int row  = (lane < 16) ? e : e + 8;
            const int colb = lane & 15;
            PsW[row * 48 + colb]      = (_Float16)p0;
            PsW[row * 48 + colb + 16] = (_Float16)p1;
        }

        // ---- PV: P(16x32) * V(32x64) ----
        FragH fp;
        {
            const int m    = lane & 15;
            const int kOff = (lane < 16) ? 0 : 8;
            fp.q[0] = *(const uint4*)(&PsW[m * 48 + kOff]);
            fp.q[1] = *(const uint4*)(&PsW[m * 48 + kOff + 16]);
        }
#pragma unroll
        for (int nb = 0; nb < 4; ++nb) {
            FragH fv;
#pragma unroll
            for (int s = 0; s < 16; ++s)
                fv.v[s] = Vs[((s < 8) ? 0 : 16) + ((lane < 16) ? 0 : 8) + (s & 7)][nb * 16 + (lane & 15)];
            acc[nb] = __builtin_amdgcn_wmma_f32_16x16x32_f16(
                false, fp.v, false, fv.v, (short)0, acc[nb], false, false);
        }
    }

    // ---- normalize + write context (f16, feeds out-proj GEMM) ----
#pragma unroll
    for (int nb = 0; nb < 4; ++nb) {
#pragma unroll
        for (int e = 0; e < 8; ++e) {
            const int row = qBlock + wid * 16 + ((lane < 16) ? e : e + 8);
            const int col = h * DHEAD + nb * 16 + (lane & 15);
            ctx[(size_t)row * DM + col] = (_Float16)(acc[nb][e] / lr[e]);
        }
    }
}

// ---------------------------------------------------------------------------
// LayerNorm over D=1024, one block per row; writes fp32 + f16 shadow.
// ---------------------------------------------------------------------------
__global__ void __launch_bounds__(256)
layernorm_k(const float* __restrict__ in, const float* __restrict__ g,
            const float* __restrict__ b, float* __restrict__ outF,
            _Float16* __restrict__ outH)
{
    __shared__ float red[256];
    const int row = blockIdx.x, t = threadIdx.x;
    float v[4];
#pragma unroll
    for (int j = 0; j < 4; ++j) v[j] = in[(size_t)row * DM + t + j * 256];
    float s = v[0] + v[1] + v[2] + v[3];
    red[t] = s; __syncthreads();
    for (int o = 128; o > 0; o >>= 1) { if (t < o) red[t] += red[t + o]; __syncthreads(); }
    const float mu = red[0] * (1.0f / DM);
    __syncthreads();
    float s2 = 0.0f;
#pragma unroll
    for (int j = 0; j < 4; ++j) { const float d = v[j] - mu; s2 += d * d; }
    red[t] = s2; __syncthreads();
    for (int o = 128; o > 0; o >>= 1) { if (t < o) red[t] += red[t + o]; __syncthreads(); }
    const float rstd = rsqrtf(red[0] * (1.0f / DM) + 1e-5f);
#pragma unroll
    for (int j = 0; j < 4; ++j) {
        const int d = t + j * 256;
        const float y = (v[j] - mu) * rstd * g[d] + b[d];
        outF[(size_t)row * DM + d] = y;
        outH[(size_t)row * DM + d] = (_Float16)y;
    }
}

// ---------------------------------------------------------------------------
// Embedding (numeric linear vs table gather) * sqrt(D) + positional encoding.
// ---------------------------------------------------------------------------
__global__ void __launch_bounds__(256)
embed_k(const float* __restrict__ values, const int* __restrict__ idx,
        const unsigned char* __restrict__ isnum, const float* __restrict__ table,
        const float* __restrict__ numw, const float* __restrict__ numb,
        float* __restrict__ xF, _Float16* __restrict__ xH)
{
    const int row = blockIdx.x, t = threadIdx.x;
    const bool numeric = isnum[row] != 0;
    const float val = values[row];
    const int   ei  = idx[row];
#pragma unroll
    for (int j = 0; j < 4; ++j) {
        const int d = t + j * 256;
        const float emb = numeric ? (val * numw[d] + numb[d])
                                  : table[(size_t)ei * DM + d];
        const float i2  = (float)((d >> 1) * 2);
        const float div = __expf(i2 * (-9.210340371976184f / 1024.0f)); // -ln(1e4)/D
        const float ang = (float)row * div;
        const float pe  = (d & 1) ? __cosf(ang) : __sinf(ang);
        const float x = emb * 32.0f + pe;                // sqrt(1024)=32
        xF[(size_t)row * DM + d] = x;
        xH[(size_t)row * DM + d] = (_Float16)x;
    }
}

// ---------------------------------------------------------------------------
// Final N=1 projection: numeric[s] = n2[s,:] . flat_w + flat_b
// ---------------------------------------------------------------------------
__global__ void __launch_bounds__(256)
flat_k(const float* __restrict__ n2, const float* __restrict__ fw,
       const float* __restrict__ fb, float* __restrict__ out)
{
    __shared__ float red[256];
    const int row = blockIdx.x, t = threadIdx.x;
    float s = 0.0f;
    for (int c = t; c < NTOK; c += 256) s += n2[(size_t)row * NTOK + c] * fw[c];
    red[t] = s; __syncthreads();
    for (int o = 128; o > 0; o >>= 1) { if (t < o) red[t] += red[t + o]; __syncthreads(); }
    if (t == 0) out[row] = red[0] + fb[0];
}

// ---------------------------------------------------------------------------
extern "C" void kernel_launch(void* const* d_in, const int* in_sizes, int n_in,
                              void* d_out, int out_size, void* d_ws, size_t ws_size,
                              hipStream_t stream)
{
    (void)in_sizes; (void)n_in; (void)out_size; (void)ws_size;

    const float*         values     = (const float*)d_in[0];
    const int*           embed_idx  = (const int*)d_in[1];
    const unsigned char* is_numeric = (const unsigned char*)d_in[2];
    const float* emb_table = (const float*)d_in[3];
    const float* num_w     = (const float*)d_in[4];
    const float* num_b     = (const float*)d_in[5];
    const float* qkv_w     = (const float*)d_in[6];
    const float* qkv_b     = (const float*)d_in[7];
    const float* out_w     = (const float*)d_in[8];
    const float* out_b     = (const float*)d_in[9];
    const float* ln1_g     = (const float*)d_in[10];
    const float* ln1_b     = (const float*)d_in[11];
    const float* lin1_w    = (const float*)d_in[12];
    const float* lin1_b    = (const float*)d_in[13];
    const float* lin2_w    = (const float*)d_in[14];
    const float* lin2_b    = (const float*)d_in[15];
    const float* ln2_g     = (const float*)d_in[16];
    const float* ln2_b     = (const float*)d_in[17];
    const float* dec_w     = (const float*)d_in[18];
    const float* dec_b     = (const float*)d_in[19];
    const float* ing_w     = (const float*)d_in[20];
    const float* ing_b     = (const float*)d_in[21];
    const float* hid_w     = (const float*)d_in[22];
    const float* hid_b     = (const float*)d_in[23];
    const float* flat_w    = (const float*)d_in[24];
    const float* flat_b    = (const float*)d_in[25];

    // workspace carve-out (~88 MB)
    char* ws = (char*)d_ws;
    size_t off = 0;
    auto carve = [&](size_t bytes) -> char* {
        char* p = ws + off;
        off += (bytes + 255) & ~(size_t)255;
        return p;
    };
    float*    xF   = (float*)   carve((size_t)SEQ * DM * 4);
    _Float16* xH   = (_Float16*)carve((size_t)SEQ * DM * 2);
    _Float16* qkvH = (_Float16*)carve((size_t)SEQ * 3 * DM * 2);
    _Float16* ctxH = (_Float16*)carve((size_t)SEQ * DM * 2);
    float*    tmpF = (float*)   carve((size_t)SEQ * DM * 4);
    _Float16* ff1H = (_Float16*)carve((size_t)SEQ * DHID * 2);
    const int N1  = 2 * NTOK;   // 4098 (true cols)
    const int N1P = 4160;       // padded to a BK=64 multiple; pad cols zeroed
    _Float16* n1H  = (_Float16*)carve((size_t)SEQ * N1P * 2);
    float*    n2F  = (float*)   carve((size_t)SEQ * NTOK * 4);

    float* outMain = (float*)d_out;                  // [S, NTOK]
    float* outNum  = outMain + (size_t)SEQ * NTOK;   // [S]

    const dim3 blk(256);
    const int gy = SEQ / 128;

    embed_k<<<SEQ, blk, 0, stream>>>(values, embed_idx, is_numeric, emb_table,
                                     num_w, num_b, xF, xH);

    for (int i = 0; i < NLAY; ++i) {
        // QKV: [S,3D] = xH * qkv_w^T + b   (f16 out only)
        gemm_wmma<<<dim3(3 * DM / 128, gy), blk, 0, stream>>>(
            xH, DM, qkv_w + (size_t)i * 3 * DM * DM, DM, DM, 3 * DM,
            qkv_b + (size_t)i * 3 * DM, nullptr, nullptr, 0, qkvH, 3 * DM,
            0, 3 * DM);
        // fused flash attention -> ctxH
        attn_wmma<<<dim3(SEQ / 128, NHEAD), blk, 0, stream>>>(qkvH, ctxH);
        // out-proj + residual -> tmpF
        gemm_wmma<<<dim3(DM / 128, gy), blk, 0, stream>>>(
            ctxH, DM, out_w + (size_t)i * DM * DM, DM, DM, DM,
            out_b + (size_t)i * DM, xF, tmpF, DM, nullptr, 0, 0, DM);
        layernorm_k<<<SEQ, blk, 0, stream>>>(tmpF, ln1_g + (size_t)i * DM,
                                             ln1_b + (size_t)i * DM, xF, xH);
        // FFN
        gemm_wmma<<<dim3(DHID / 128, gy), blk, 0, stream>>>(
            xH, DM, lin1_w + (size_t)i * DHID * DM, DM, DM, DHID,
            lin1_b + (size_t)i * DHID, nullptr, nullptr, 0, ff1H, DHID,
            1, DHID);
        gemm_wmma<<<dim3(DM / 128, gy), blk, 0, stream>>>(
            ff1H, DHID, lin2_w + (size_t)i * DM * DHID, DHID, DHID, DM,
            lin2_b + (size_t)i * DM, xF, tmpF, DM, nullptr, 0, 0, DM);
        layernorm_k<<<SEQ, blk, 0, stream>>>(tmpF, ln2_g + (size_t)i * DM,
                                             ln2_b + (size_t)i * DM, xF, xH);
    }

    // head: numeric path (n1H padded 4098 -> 4160 with zeros so the hid GEMM
    // K-loop stays a multiple of 64 with no scalar remainder path)
    gemm_wmma<<<dim3((N1P + 127) / 128, gy), blk, 0, stream>>>(
        xH, DM, ing_w, DM, DM, N1, ing_b, nullptr, nullptr, 0, n1H, N1P,
        1, N1P);
    gemm_wmma<<<dim3((NTOK + 127) / 128, gy), blk, 0, stream>>>(
        n1H, N1P, hid_w, N1P, N1, NTOK, hid_b, nullptr, n2F, NTOK,
        nullptr, 0, 1, NTOK);
    flat_k<<<SEQ, blk, 0, stream>>>(n2F, flat_w, flat_b, outNum);

    // head: decode path -> d_out[0 : S*NTOK]
    gemm_wmma<<<dim3((NTOK + 127) / 128, gy), blk, 0, stream>>>(
        xH, DM, dec_w, DM, DM, NTOK, dec_b, nullptr, outMain, NTOK,
        nullptr, 0, 0, NTOK);
}
